// GAT_41832981463079
// MI455X (gfx1250) — matrix operand order repaired and checked by reading, not compiled, not run on hardware
//
#include <hip/hip_runtime.h>
#include <hip/hip_bf16.h>
#include <cstdint>

#define GAT_NEG_SLOPE 0.2f

typedef __attribute__((ext_vector_type(16))) __bf16          v16bf;
typedef __attribute__((ext_vector_type(16))) unsigned short  v16u16;
typedef __attribute__((ext_vector_type(8)))  float           v8f;

// ---------------- helpers ----------------

__device__ inline unsigned short f2bf(float f) {           // RNE f32 -> bf16
    unsigned u = __float_as_uint(f);
    u += 0x7FFFu + ((u >> 16) & 1u);
    return (unsigned short)(u >> 16);
}

// monotonic unsigned key for float atomic-max (handles negatives); init key = 0
__device__ inline unsigned fkey(float f) {
    unsigned u = __float_as_uint(f);
    return (u & 0x80000000u) ? ~u : (u | 0x80000000u);
}
__device__ inline float fdec(unsigned e) {
    unsigned u = (e & 0x80000000u) ? (e ^ 0x80000000u) : ~e;
    return __uint_as_float(u);
}

__device__ inline float lrelu(float v) { return v > 0.f ? v : GAT_NEG_SLOPE * v; }

// ---------------- operand packing into wave32 WMMA layouts (f32 -> bf16) ----------------

// A-matrix 16x32 bf16 tiles: lane L holds row m=L&15; half=L>>4 selects K runs
// {0..7,16..23} (half 0) or {8..15,24..31} (half 1). One thread = one lane slot.
__global__ void pack_a_kernel(const float* __restrict__ X, unsigned short* __restrict__ P,
                              int Mt, int Kt, int ld, const float* __restrict__ bias, int elu)
{
    int idx = blockIdx.x * blockDim.x + threadIdx.x;
    int total = Mt * Kt * 32;
    if (idx >= total) return;
    int lane = idx & 31;
    int t    = idx >> 5;
    int kt   = t % Kt;
    int mt   = t / Kt;
    int m    = lane & 15;
    int half = lane >> 4;
    int col0 = kt * 32 + half * 8;
    const float* row = X + (size_t)(mt * 16 + m) * ld;
    unsigned short* o = P + (size_t)idx * 16;
#pragma unroll
    for (int i = 0; i < 8; i++) {
        float v = row[col0 + i];
        if (bias) v += bias[col0 + i];
        if (elu)  v = v > 0.f ? v : (__expf(v) - 1.f);
        o[i] = f2bf(v);
    }
#pragma unroll
    for (int i = 0; i < 8; i++) {
        float v = row[col0 + 16 + i];
        if (bias) v += bias[col0 + 16 + i];
        if (elu)  v = v > 0.f ? v : (__expf(v) - 1.f);
        o[8 + i] = f2bf(v);
    }
}

// B-matrix 32x16 bf16 tiles: lane L holds column n=L&15; half selects K runs {0..7,16..23} / {8..15,24..31}.
__global__ void pack_b_kernel(const float* __restrict__ X, unsigned short* __restrict__ P,
                              int Kt, int Nt, int ld)
{
    int idx = blockIdx.x * blockDim.x + threadIdx.x;
    int total = Kt * Nt * 32;
    if (idx >= total) return;
    int lane = idx & 31;
    int t    = idx >> 5;
    int nt   = t % Nt;
    int kt   = t / Nt;
    int n    = lane & 15;
    int half = lane >> 4;
    const float* base = X + (size_t)(kt * 32 + half * 8) * ld + nt * 16 + n;
    unsigned short* o = P + (size_t)idx * 16;
#pragma unroll
    for (int i = 0; i < 16; i++) o[i] = f2bf(base[(size_t)(i < 8 ? i : i + 8) * ld]);
}

// ---------------- WMMA GEMM: one 16x16 f32 tile per wave ----------------

__global__ __launch_bounds__(256) void wmma_gemm_kernel(
    const unsigned short* __restrict__ Ap, const unsigned short* __restrict__ Bp,
    float* __restrict__ C, int Mt, int Nt, int Kt, int ldc)
{
    int wid  = (blockIdx.x * blockDim.x + threadIdx.x) >> 5;
    int lane = threadIdx.x & 31;
    if (wid >= Mt * Nt) return;          // whole-wave uniform exit (EXEC all-1 for WMMA)
    int mt = wid / Nt, nt = wid % Nt;
    const v16u16* a = (const v16u16*)Ap + (size_t)mt * Kt * 32 + lane;
    const v16u16* b = (const v16u16*)Bp + (size_t)nt * 32 + lane;
    v8f acc = {};
    for (int k = 0; k < Kt; k++) {
        v16u16 av = a[(size_t)k * 32];
        v16u16 bv = b[(size_t)k * Nt * 32];
        acc = __builtin_amdgcn_wmma_f32_16x16x32_bf16(
            false, __builtin_bit_cast(v16bf, av),
            false, __builtin_bit_cast(v16bf, bv),
            (short)0, acc, false, false);
    }
    int half = lane >> 4, col = lane & 15;
    float* c = C + (size_t)(mt * 16 + half * 8) * ldc + nt * 16 + col;
#pragma unroll
    for (int r = 0; r < 8; r++) c[(size_t)r * ldc] = acc[r];
}

// ---------------- attention logits: al[n,h] = <h[n,h,:], att[h,:]> ----------------

__global__ void att_logits_kernel(const float* __restrict__ h,
                                  const float* __restrict__ asrc, const float* __restrict__ adst,
                                  float* __restrict__ als, float* __restrict__ ald,
                                  int N, int H, int C)
{
    int idx = blockIdx.x * blockDim.x + threadIdx.x;
    if (idx >= N * H) return;
    int n = idx / H, hh = idx % H;
    const float* p = h + (size_t)n * H * C + (size_t)hh * C;
    float s = 0.f, d = 0.f;
    for (int c = 0; c < C; c++) {
        float v = p[c];
        s += v * asrc[hh * C + c];
        d += v * adst[hh * C + c];
    }
    als[idx] = s;
    ald[idx] = d;
}

// ---------------- init kernels ----------------

__global__ void fill_f32_kernel(float* __restrict__ p, int n, float v) {
    int i = blockIdx.x * blockDim.x + threadIdx.x;
    if (i < n) p[i] = v;
}
__global__ void fill_u32_kernel(unsigned* __restrict__ p, int n, unsigned v) {
    int i = blockIdx.x * blockDim.x + threadIdx.x;
    if (i < n) p[i] = v;
}
__global__ void init_out_kernel(float* __restrict__ out, const float* __restrict__ b, int N, int C) {
    int i = blockIdx.x * blockDim.x + threadIdx.x;
    if (i < N * C) out[i] = b[i % C];
}

// ---------------- edge softmax passes (with self loops appended) ----------------

__global__ void edge_max_kernel(const int* __restrict__ ei, int E, int Etot,
                                const float* __restrict__ als, const float* __restrict__ ald,
                                unsigned* __restrict__ mkey, int H)
{
    int e = blockIdx.x * blockDim.x + threadIdx.x;
    if (e >= Etot) return;
    int s = (e < E) ? ei[e]     : (e - E);
    int d = (e < E) ? ei[E + e] : (e - E);
    for (int h = 0; h < H; h++) {
        float v = lrelu(als[s * H + h] + ald[d * H + h]);
        atomicMax(&mkey[d * H + h], fkey(v));
    }
}

__global__ void edge_sum_kernel(const int* __restrict__ ei, int E, int Etot,
                                const float* __restrict__ als, const float* __restrict__ ald,
                                const unsigned* __restrict__ mkey, float* __restrict__ ssum, int H)
{
    int e = blockIdx.x * blockDim.x + threadIdx.x;
    if (e >= Etot) return;
    int s = (e < E) ? ei[e]     : (e - E);
    int d = (e < E) ? ei[E + e] : (e - E);
    for (int h = 0; h < H; h++) {
        float v = lrelu(als[s * H + h] + ald[d * H + h]);
        atomicAdd(&ssum[d * H + h], __expf(v - fdec(mkey[d * H + h])));
    }
}

// one wave per edge; per-head alpha in lanes 0..H-1, broadcast via shuffle
__global__ __launch_bounds__(256) void edge_agg_kernel(
    const int* __restrict__ ei, int E, int Etot,
    const float* __restrict__ als, const float* __restrict__ ald,
    const unsigned* __restrict__ mkey, const float* __restrict__ ssum,
    const float* __restrict__ hsrc, float* __restrict__ agg, int H, int C)
{
    int wid  = (blockIdx.x * blockDim.x + threadIdx.x) >> 5;
    int lane = threadIdx.x & 31;
    if (wid >= Etot) return;
    int e = wid;
    int s = (e < E) ? ei[e]     : (e - E);
    int d = (e < E) ? ei[E + e] : (e - E);
    float alpha = 0.f;
    if (lane < H) {
        float v = lrelu(als[s * H + lane] + ald[d * H + lane]);
        alpha = __expf(v - fdec(mkey[d * H + lane])) / ssum[d * H + lane];
    }
    int HC = H * C;
    const float* hp = hsrc + (size_t)s * HC;
    float* ap       = agg  + (size_t)d * HC;
    for (int i = lane; i < HC; i += 32) {
        float a = __shfl(alpha, i / C, 32);
        atomicAdd(&ap[i], hp[i] * a);
    }
}

// ---------------- host orchestration ----------------

extern "C" void kernel_launch(void* const* d_in, const int* in_sizes, int n_in,
                              void* d_out, int out_size, void* d_ws, size_t ws_size,
                              hipStream_t stream)
{
    const float* x   = (const float*)d_in[0];
    const int*   ei  = (const int*)  d_in[1];
    const float* W1  = (const float*)d_in[2];
    const float* as1 = (const float*)d_in[3];
    const float* ad1 = (const float*)d_in[4];
    const float* b1  = (const float*)d_in[5];
    const float* W2  = (const float*)d_in[6];
    const float* as2 = (const float*)d_in[7];
    const float* ad2 = (const float*)d_in[8];
    const float* b2  = (const float*)d_in[9];
    float* out = (float*)d_out;
    (void)n_in; (void)out_size; (void)ws_size;

    const int F_IN = 256, H1 = 8, HID = 64, OUTC = 64;
    const int HC1 = H1 * HID;                 // 512
    const int N = in_sizes[0] / F_IN;         // 50000 (multiple of 16)
    const int E = in_sizes[1] / 2;            // 800000
    const int Etot = E + N;                   // + self loops
    const int Mt  = N / 16;                   // 3125
    const int Kt1 = F_IN / 32;                // 8
    const int Nt1 = HC1 / 16;                 // 32
    const int Kt2 = HC1 / 32;                 // 16
    const int Nt2 = OUTC / 16;                // 4

    char* ws = (char*)d_ws;
    size_t off = 0;
    auto alloc = [&](size_t bytes) -> void* {
        void* p = ws + off;
        off += (bytes + 255) & ~(size_t)255;
        return p;
    };
    unsigned short* Ap   = (unsigned short*)alloc((size_t)N * HC1 * 2);  // reused for both layers
    unsigned short* Bp   = (unsigned short*)alloc((size_t)F_IN * HC1 * 2);
    float*    h1   = (float*)   alloc((size_t)N * HC1 * 4);
    float*    agg1 = (float*)   alloc((size_t)N * HC1 * 4);
    float*    als1 = (float*)   alloc((size_t)N * H1 * 4);
    float*    ald1 = (float*)   alloc((size_t)N * H1 * 4);
    unsigned* m1   = (unsigned*)alloc((size_t)N * H1 * 4);
    float*    s1   = (float*)   alloc((size_t)N * H1 * 4);
    float*    h2   = (float*)   alloc((size_t)N * OUTC * 4);
    float*    als2 = (float*)   alloc((size_t)N * 4);
    float*    ald2 = (float*)   alloc((size_t)N * 4);
    unsigned* m2   = (unsigned*)alloc((size_t)N * 4);
    float*    s2   = (float*)   alloc((size_t)N * 4);

    const int B = 256;
    auto cdiv = [](long long a, long long b) { return (int)((a + b - 1) / b); };

    // ---- Layer 1 ----
    pack_a_kernel<<<cdiv((long long)Mt * Kt1 * 32, B), B, 0, stream>>>(x, Ap, Mt, Kt1, F_IN, nullptr, 0);
    pack_b_kernel<<<cdiv((long long)Kt1 * Nt1 * 32, B), B, 0, stream>>>(W1, Bp, Kt1, Nt1, HC1);
    wmma_gemm_kernel<<<cdiv((long long)Mt * Nt1 * 32, B), B, 0, stream>>>(Ap, Bp, h1, Mt, Nt1, Kt1, HC1);
    att_logits_kernel<<<cdiv((long long)N * H1, B), B, 0, stream>>>(h1, as1, ad1, als1, ald1, N, H1, HID);
    fill_u32_kernel<<<cdiv((long long)N * H1, B), B, 0, stream>>>(m1, N * H1, 0u);
    fill_f32_kernel<<<cdiv((long long)N * H1, B), B, 0, stream>>>(s1, N * H1, 0.f);
    fill_f32_kernel<<<cdiv((long long)N * HC1, B), B, 0, stream>>>(agg1, N * HC1, 0.f);
    edge_max_kernel<<<cdiv(Etot, B), B, 0, stream>>>(ei, E, Etot, als1, ald1, m1, H1);
    edge_sum_kernel<<<cdiv(Etot, B), B, 0, stream>>>(ei, E, Etot, als1, ald1, m1, s1, H1);
    edge_agg_kernel<<<cdiv((long long)Etot * 32, B), B, 0, stream>>>(ei, E, Etot, als1, ald1, m1, s1, h1, agg1, H1, HID);

    // ---- Layer 2 (ELU(agg1 + b1) folded into A-pack) ----
    pack_a_kernel<<<cdiv((long long)Mt * Kt2 * 32, B), B, 0, stream>>>(agg1, Ap, Mt, Kt2, HC1, b1, 1);
    pack_b_kernel<<<cdiv((long long)Kt2 * Nt2 * 32, B), B, 0, stream>>>(W2, Bp, Kt2, Nt2, OUTC);
    wmma_gemm_kernel<<<cdiv((long long)Mt * Nt2 * 32, B), B, 0, stream>>>(Ap, Bp, h2, Mt, Nt2, Kt2, OUTC);
    att_logits_kernel<<<cdiv((long long)N, B), B, 0, stream>>>(h2, as2, ad2, als2, ald2, N, 1, OUTC);
    fill_u32_kernel<<<cdiv((long long)N, B), B, 0, stream>>>(m2, N, 0u);
    fill_f32_kernel<<<cdiv((long long)N, B), B, 0, stream>>>(s2, N, 0.f);
    init_out_kernel<<<cdiv((long long)N * OUTC, B), B, 0, stream>>>(out, b2, N, OUTC);
    edge_max_kernel<<<cdiv(Etot, B), B, 0, stream>>>(ei, E, Etot, als2, ald2, m2, 1);
    edge_sum_kernel<<<cdiv(Etot, B), B, 0, stream>>>(ei, E, Etot, als2, ald2, m2, s2, 1);
    edge_agg_kernel<<<cdiv((long long)Etot * 32, B), B, 0, stream>>>(ei, E, Etot, als2, ald2, m2, s2, h2, out, 1, OUTC);
}
